// HebbianLinear_55697135895031
// MI455X (gfx1250) — matrix operand order repaired and checked by reading, not compiled, not run on hardware
//
#include <hip/hip_runtime.h>
#include <math.h>

// ---------------------------------------------------------------------------
// Softhebb step for MI455X (gfx1250), bf16 WMMA path, software-pipelined.
// B=4096, IN=2048, OUT=2048.
// ---------------------------------------------------------------------------

typedef __bf16 bf16;
typedef __bf16 v16bf __attribute__((ext_vector_type(16)));
typedef __bf16 v8bf  __attribute__((ext_vector_type(8)));
typedef float  v8f   __attribute__((ext_vector_type(8)));
typedef float  v4f   __attribute__((ext_vector_type(4)));

#define B_DIM   4096
#define IN_DIM  2048
#define OUT_DIM 2048
#define RATE_C  0.01f
#define INV_B   (1.0f / 4096.0f)

// ---- A-fragment loader -----------------------------------------------------
// 16-bit A matrix 16x32 layout (ISA 7.12.2): lanes 0-15 hold row M=lane with
// K = {0..7, 16..23}; lanes 16-31 hold row M=lane-16 with K = {8..15, 24..31}.
// p = row_base + khalf*8; fetch 16-byte chunks at +0 and +16 elements.
static __device__ __forceinline__ v16bf load_a_frag(const bf16* p) {
  v8bf lo = *reinterpret_cast<const v8bf*>(p);
  v8bf hi = *reinterpret_cast<const v8bf*>(p + 16);
  v16bf f;
#pragma unroll
  for (int i = 0; i < 8; ++i) { f[i] = lo[i]; f[i + 8] = hi[i]; }
  return f;
}

// ---- WMMA 64x32 tile core, register double-buffered over K -----------------
template <int KDIM>
static __device__ __forceinline__ void wmma_tile_64x32(
    const bf16* __restrict__ A, const bf16* __restrict__ Bm,
    int M0, int N0, int lane, v8f acc[4][2]) {
  const int hrow  = lane & 15;
  const int khalf = lane >> 4;

  const bf16* aptr[4];
#pragma unroll
  for (int mt = 0; mt < 4; ++mt)
    aptr[mt] = A + (size_t)(M0 + mt * 16 + hrow) * KDIM + khalf * 8;
  const bf16* bptr[2];
#pragma unroll
  for (int nt = 0; nt < 2; ++nt)
    bptr[nt] = Bm + (size_t)(N0 + nt * 16 + hrow) * KDIM + khalf * 8;

  v16bf abuf[2][4], bbuf[2][2];

  // prologue: fragments for kk = 0
#pragma unroll
  for (int mt = 0; mt < 4; ++mt) abuf[0][mt] = load_a_frag(aptr[mt]);
#pragma unroll
  for (int nt = 0; nt < 2; ++nt)
    bbuf[0][nt] = load_a_frag(bptr[nt]);

  int cur = 0;
#pragma unroll 2
  for (int kk = 32; kk < KDIM; kk += 32) {
    const int nxt = cur ^ 1;
    __builtin_prefetch(aptr[0] + kk + 512, 0, 1);   // global_prefetch_b8
    __builtin_prefetch(bptr[0] + kk + 512, 0, 1);
    // issue next k-block loads first, then compute on current fragments
#pragma unroll
    for (int mt = 0; mt < 4; ++mt) abuf[nxt][mt] = load_a_frag(aptr[mt] + kk);
#pragma unroll
    for (int nt = 0; nt < 2; ++nt)
      bbuf[nxt][nt] = load_a_frag(bptr[nt] + kk);
#pragma unroll
    for (int mt = 0; mt < 4; ++mt)
#pragma unroll
      for (int nt = 0; nt < 2; ++nt)
        acc[mt][nt] = __builtin_amdgcn_wmma_f32_16x16x32_bf16(
            false, abuf[cur][mt], false, bbuf[cur][nt], (short)0,
            acc[mt][nt], false, false);
    cur = nxt;
  }
  // epilogue compute on last fragments
#pragma unroll
  for (int mt = 0; mt < 4; ++mt)
#pragma unroll
    for (int nt = 0; nt < 2; ++nt)
      acc[mt][nt] = __builtin_amdgcn_wmma_f32_16x16x32_bf16(
          false, abuf[cur][mt], false, bbuf[cur][nt], (short)0,
          acc[mt][nt], false, false);
}

// ---- elementwise f32 -> bf16, 8 elems/thread -------------------------------
__global__ __launch_bounds__(256)
void cvt_bf16_kernel(const float* __restrict__ src, bf16* __restrict__ dst, int n8) {
  const int i = blockIdx.x * 256 + threadIdx.x;
  if (i >= n8) return;
  const v4f* s4 = reinterpret_cast<const v4f*>(src);
  v4f lo = s4[2 * i], hi = s4[2 * i + 1];
  v8bf d;
#pragma unroll
  for (int j = 0; j < 4; ++j) { d[j] = (bf16)lo[j]; d[4 + j] = (bf16)hi[j]; }
  reinterpret_cast<v8bf*>(dst)[i] = d;
}

// ---- LDS-tiled transpose + convert: [rows, cols] f32 -> [cols, rows] bf16 --
__global__ __launch_bounds__(256)
void transpose_cvt_kernel(const float* __restrict__ src, bf16* __restrict__ dst,
                          int rows, int cols) {
  __shared__ bf16 tile[32][33];
  const int tx = threadIdx.x & 31;
  const int ty = threadIdx.x >> 5;
  const int c0 = blockIdx.x * 32;
  const int r0 = blockIdx.y * 32;
#pragma unroll
  for (int rr = ty; rr < 32; rr += 8)
    tile[rr][tx] = (bf16)src[(size_t)(r0 + rr) * cols + (c0 + tx)];
  __syncthreads();
#pragma unroll
  for (int rr = ty; rr < 32; rr += 8)
    dst[(size_t)(c0 + rr) * rows + (r0 + tx)] = tile[tx][rr];
}

// ---- GEMM1: U[b,o] = sum_k xbf[b,k]*wbf[o,k] + bias[o] ---------------------
__global__ __launch_bounds__(256)
void gemm_u_kernel(const bf16* __restrict__ A, const bf16* __restrict__ Bm,
                   const float* __restrict__ bias, float* __restrict__ U) {
  const int lane  = threadIdx.x & 31;
  const int wave  = threadIdx.x >> 5;
  const int M0 = blockIdx.y * 128 + (wave >> 2) * 64;   // batch rows
  const int N0 = blockIdx.x * 128 + (wave & 3) * 32;    // output cols

  v8f acc[4][2] = {};
  wmma_tile_64x32<IN_DIM>(A, Bm, M0, N0, lane, acc);

  // C/D layout: lane<16 -> N=lane, VGPR r -> M=r; lane>=16 -> N=lane-16, M=8+r.
  const int ncol  = lane & 15;
  const int chalf = lane >> 4;
#pragma unroll
  for (int nt = 0; nt < 2; ++nt) {
    const int col = N0 + nt * 16 + ncol;
    const float bz = bias[col];
#pragma unroll
    for (int mt = 0; mt < 4; ++mt)
#pragma unroll
      for (int r = 0; r < 8; ++r) {
        const int row = M0 + mt * 16 + chalf * 8 + r;
        U[(size_t)row * OUT_DIM + col] = acc[mt][nt][r] + bz;
      }
  }
}

// ---- row softmax over U (single global read), store y^T bf16 ---------------
__global__ __launch_bounds__(256)
void softmax_transpose_kernel(const float* __restrict__ U, bf16* __restrict__ Yt) {
  __shared__ float red[256];
  const int tid = threadIdx.x;
  const int b = blockIdx.x;
  const float* urow = U + (size_t)b * OUT_DIM;

  float v[8];                                  // OUT_DIM / 256 == 8
#pragma unroll
  for (int t = 0; t < 8; ++t) v[t] = urow[tid + t * 256];

  float mx = v[0];
#pragma unroll
  for (int t = 1; t < 8; ++t) mx = fmaxf(mx, v[t]);
  red[tid] = mx; __syncthreads();
  for (int s = 128; s > 0; s >>= 1) {
    if (tid < s) red[tid] = fmaxf(red[tid], red[tid + s]);
    __syncthreads();
  }
  const float rowmax = red[0];
  __syncthreads();

  float sum = 0.0f;
#pragma unroll
  for (int t = 0; t < 8; ++t) { v[t] = __expf(v[t] - rowmax); sum += v[t]; }
  red[tid] = sum; __syncthreads();
  for (int s = 128; s > 0; s >>= 1) {
    if (tid < s) red[tid] += red[tid + s];
    __syncthreads();
  }
  const float inv = 1.0f / red[0];

#pragma unroll
  for (int t = 0; t < 8; ++t)
    Yt[(size_t)(tid + t * 256) * B_DIM + b] = (bf16)(v[t] * inv);   // TEMP == 1
}

// ---- yu[o] = sum_b y[b,o]*u[b,o]  (one wave32 per o) -----------------------
__global__ __launch_bounds__(256)
void col_yu_kernel(const bf16* __restrict__ Yt, const float* __restrict__ U,
                   float* __restrict__ yu) {
  const int lane = threadIdx.x & 31;
  const int o = blockIdx.x * 8 + (threadIdx.x >> 5);
  const bf16* yrow = Yt + (size_t)o * B_DIM;
  float s = 0.0f;
  for (int b = lane; b < B_DIM; b += 32)
    s += (float)yrow[b] * U[(size_t)b * OUT_DIM + o];
#pragma unroll
  for (int off = 16; off > 0; off >>= 1) s += __shfl_xor(s, off, 32);
  if (lane == 0) yu[o] = s;
}

// ---- rate[o] = RATE * |1 - ||W_row|||^0.5  (one wave32 per o) --------------
__global__ __launch_bounds__(256)
void row_rate_kernel(const float* __restrict__ W, float* __restrict__ rate) {
  const int lane = threadIdx.x & 31;
  const int o = blockIdx.x * 8 + (threadIdx.x >> 5);
  const float* wrow = W + (size_t)o * IN_DIM;
  float ss = 0.0f;
  for (int i = lane; i < IN_DIM; i += 32) { const float w = wrow[i]; ss += w * w; }
#pragma unroll
  for (int off = 16; off > 0; off >>= 1) ss += __shfl_xor(ss, off, 32);
  if (lane == 0) rate[o] = RATE_C * sqrtf(fabsf(1.0f - sqrtf(ss)));
}

// ---- GEMM2 + fused epilogue ------------------------------------------------
// yx[o,i] = sum_b yT[o,b]*xT[i,b];  out = rate[o]*(yx - yu[o]*W[o,i])/B
__global__ __launch_bounds__(256)
void gemm_step_kernel(const bf16* __restrict__ A /* yT [OUT x B] */,
                      const bf16* __restrict__ Bm /* xT [IN x B] */,
                      const float* __restrict__ W, const float* __restrict__ yu,
                      const float* __restrict__ rate, float* __restrict__ out) {
  const int lane  = threadIdx.x & 31;
  const int wave  = threadIdx.x >> 5;
  const int M0 = blockIdx.y * 128 + (wave >> 2) * 64;   // output rows o
  const int N0 = blockIdx.x * 128 + (wave & 3) * 32;    // input cols i

  v8f acc[4][2] = {};
  wmma_tile_64x32<B_DIM>(A, Bm, M0, N0, lane, acc);

  const int ncol  = lane & 15;
  const int chalf = lane >> 4;
#pragma unroll
  for (int mt = 0; mt < 4; ++mt)
#pragma unroll
    for (int r = 0; r < 8; ++r) {
      const int o = M0 + mt * 16 + chalf * 8 + r;
      const float sc  = rate[o] * INV_B;
      const float yuo = yu[o];
#pragma unroll
      for (int nt = 0; nt < 2; ++nt) {
        const int i = N0 + nt * 16 + ncol;
        out[(size_t)o * IN_DIM + i] =
            sc * (acc[mt][nt][r] - yuo * W[(size_t)o * IN_DIM + i]);
      }
    }
}

// ---------------------------------------------------------------------------
extern "C" void kernel_launch(void* const* d_in, const int* in_sizes, int n_in,
                              void* d_out, int out_size, void* d_ws, size_t ws_size,
                              hipStream_t stream) {
  const float* x    = (const float*)d_in[0];  // [B, IN]
  const float* w    = (const float*)d_in[1];  // [OUT, IN]
  const float* bias = (const float*)d_in[2];  // [OUT]
  float* out = (float*)d_out;                 // [OUT, IN]

  // Workspace layout (~88 MB total)
  char* ws = (char*)d_ws;
  float* U   = (float*)ws;  ws += (size_t)B_DIM * OUT_DIM * sizeof(float);   // 32 MB
  bf16* xbf  = (bf16*)ws;   ws += (size_t)B_DIM * IN_DIM * sizeof(bf16);     // 16 MB
  bf16* wbf  = (bf16*)ws;   ws += (size_t)OUT_DIM * IN_DIM * sizeof(bf16);   //  8 MB
  bf16* xt   = (bf16*)ws;   ws += (size_t)IN_DIM * B_DIM * sizeof(bf16);     // 16 MB
  bf16* yt   = (bf16*)ws;   ws += (size_t)OUT_DIM * B_DIM * sizeof(bf16);    // 16 MB
  float* yu  = (float*)ws;  ws += (size_t)OUT_DIM * sizeof(float);
  float* rate = (float*)ws;

  // 1. convert operands to bf16 (8 elems/thread)
  cvt_bf16_kernel<<<(B_DIM * IN_DIM / 8) / 256, 256, 0, stream>>>(x, xbf,
                                                                  B_DIM * IN_DIM / 8);
  cvt_bf16_kernel<<<(OUT_DIM * IN_DIM / 8) / 256, 256, 0, stream>>>(w, wbf,
                                                                    OUT_DIM * IN_DIM / 8);
  // 2. xT (bf16) for GEMM2's B operand
  {
    dim3 g(IN_DIM / 32, B_DIM / 32);
    transpose_cvt_kernel<<<g, 256, 0, stream>>>(x, xt, B_DIM, IN_DIM);
  }
  // 3. u = x @ W^T + bias
  {
    dim3 g(OUT_DIM / 128, B_DIM / 128);
    gemm_u_kernel<<<g, 256, 0, stream>>>(xbf, wbf, bias, U);
  }
  // 4. y = softmax(u) rows, stored transposed bf16
  softmax_transpose_kernel<<<B_DIM, 256, 0, stream>>>(U, yt);
  // 5. yu[o], rate[o]
  col_yu_kernel<<<OUT_DIM / 8, 256, 0, stream>>>(yt, U, yu);
  row_rate_kernel<<<OUT_DIM / 8, 256, 0, stream>>>(w, rate);
  // 6. yx = y^T @ x, fused epilogue -> step
  {
    dim3 g(IN_DIM / 128, OUT_DIM / 128);
    gemm_step_kernel<<<g, 256, 0, stream>>>(yt, xt, w, yu, rate, out);
  }
}
